// FusedMoE_38431367365169
// MI455X (gfx1250) — compile-verified, matrix operand order checked by reading, not run on hardware
//
#include <hip/hip_runtime.h>

#define T_TOK 8192
#define HID   2048
#define INTER 4096
#define NEXP  8

typedef __attribute__((ext_vector_type(16))) __bf16 v16bf;
typedef __attribute__((ext_vector_type(8)))  float  v8f;
typedef unsigned short u16;

union FragB16 { uint4 q[2]; v16bf v; };
union Pack4   { __bf16 b[4]; uint2 u; };

static __device__ __forceinline__ v16bf ldsFrag(const __bf16* base, int row, int khalf) {
    FragB16 f;
    const __bf16* p = base + row * 40;
    f.q[0] = *(const uint4*)(p + khalf);        // K in [khalf, khalf+8)
    f.q[1] = *(const uint4*)(p + khalf + 16);   // K in [khalf+16, khalf+24)
    return f.v;
}

static __device__ __forceinline__ void cvtStore4(__bf16* d, float4 f) {
    d[0] = (__bf16)f.x; d[1] = (__bf16)f.y; d[2] = (__bf16)f.z; d[3] = (__bf16)f.w;
}

// ---------------- router: softmax top-2 + scatter to expert buckets ----------------
__global__ void moe_router(const float* __restrict__ logits,
                           int* __restrict__ cnt, int* __restrict__ tok_ids,
                           float* __restrict__ tok_w) {
    int t = blockIdx.x * blockDim.x + threadIdx.x;
    if (t >= T_TOK) return;
    float l[NEXP];
#pragma unroll
    for (int j = 0; j < NEXP; ++j) l[j] = logits[t * NEXP + j];
    int i0 = 0; float m0 = l[0];
#pragma unroll
    for (int j = 1; j < NEXP; ++j) if (l[j] > m0) { m0 = l[j]; i0 = j; }
    int i1 = (i0 == 0) ? 1 : 0; float m1 = l[i1];
#pragma unroll
    for (int j = 0; j < NEXP; ++j)
        if (j != i0 && l[j] > m1) { m1 = l[j]; i1 = j; }
    float e1 = __expf(m1 - m0);
    float w0 = 1.0f / (1.0f + e1);
    float w1 = 1.0f - w0;
    int p0 = atomicAdd(&cnt[i0], 1);
    tok_ids[i0 * T_TOK + p0] = t;  tok_w[i0 * T_TOK + p0] = w0;
    int p1 = atomicAdd(&cnt[i1], 1);
    tok_ids[i1 * T_TOK + p1] = t;  tok_w[i1 * T_TOK + p1] = w1;
}

__global__ void scan_counts(const int* __restrict__ cnt, int* __restrict__ off) {
    if (threadIdx.x == 0) {
        int s = 0;
        for (int e = 0; e < NEXP; ++e) { off[e] = s; s += cnt[e]; }
    }
}

// ---------------- x fp32 -> bf16 copy ----------------
__global__ void cvt_x(const float* __restrict__ x, u16* __restrict__ xb, int n4) {
    int i = blockIdx.x * blockDim.x + threadIdx.x;
    if (i >= n4) return;
    float4 f = ((const float4*)x)[i];
    Pack4 p;
    p.b[0] = (__bf16)f.x; p.b[1] = (__bf16)f.y; p.b[2] = (__bf16)f.z; p.b[3] = (__bf16)f.w;
    ((uint2*)xb)[i] = p.u;
}

// ---------------- GEMM1: act = silu(x@Wg^T) * (x@Wu^T), bf16 out ----------------
__global__ __launch_bounds__(256)
void moe_gemm1(const u16* __restrict__ xb_u, const float* __restrict__ w13,
               const int* __restrict__ cnt, const int* __restrict__ off,
               const int* __restrict__ tok_ids, u16* __restrict__ act_u) {
    const int e  = blockIdx.z;
    const int n0 = blockIdx.x * 128;   // intermediate-dim tile
    const int s0 = blockIdx.y * 128;   // slot tile
    const int c  = cnt[e];
    if (s0 >= c) return;
    const int base = off[e];
    const __bf16* xb = (const __bf16*)xb_u;
    __bf16* act = (__bf16*)act_u;

    __shared__ alignas(16) __bf16 As[128 * 40];
    __shared__ alignas(16) __bf16 Bg[128 * 40];
    __shared__ alignas(16) __bf16 Bu[128 * 40];
    __shared__ int ts[128];

    const int tid = threadIdx.x;
    if (tid < 128) {
        int slot = s0 + tid;
        int sc = slot < c ? slot : (c - 1);
        ts[tid] = tok_ids[e * T_TOK + sc];
    }
    __syncthreads();

    // ---- loop-invariant streaming pointers (4 chunks per thread) ----
    const size_t wgate = ((size_t)e * 2 * INTER + (size_t)n0) * HID;
    const size_t wup   = wgate + (size_t)INTER * HID;
    const __bf16* pa[4]; const float* pg[4]; const float* pu[4]; int lo[4];
#pragma unroll
    for (int i = 0; i < 4; ++i) {
        int cidx = tid + 256 * i;
        int row  = cidx >> 3;
        int kc   = (cidx & 7) * 4;
        pa[i] = xb + (size_t)ts[row] * HID + kc;
        pg[i] = w13 + wgate + (size_t)row * HID + kc;
        pu[i] = w13 + wup   + (size_t)row * HID + kc;
        lo[i] = row * 40 + kc;
    }

    const int lane  = tid & 31;
    const int wv    = tid >> 5;
    const int m0w   = (wv & 3) * 32;   // 4 waves over 128 rows
    const int n0w   = (wv >> 2) * 64;  // 2 waves over 128 cols
    const int lrow  = lane & 15;
    const int khalf = (lane >> 4) << 3;

    v8f accg[2][4], accu[2][4];
#pragma unroll
    for (int mi = 0; mi < 2; ++mi)
#pragma unroll
        for (int ni = 0; ni < 4; ++ni) { accg[mi][ni] = {}; accu[mi][ni] = {}; }

    // ---- prologue: stage tile k=0 into registers ----
    uint2 ar[4]; float4 gr[4], ur[4];
#pragma unroll
    for (int i = 0; i < 4; ++i) {
        ar[i] = *(const uint2*)pa[i];
        gr[i] = *(const float4*)pg[i];
        ur[i] = *(const float4*)pu[i];
        pa[i] += 32; pg[i] += 32; pu[i] += 32;
    }

    for (int k0 = 0; k0 < HID; k0 += 32) {
        __syncthreads();   // previous iteration's fragment reads are done
#pragma unroll
        for (int i = 0; i < 4; ++i) {
            *(uint2*)(As + lo[i]) = ar[i];
            cvtStore4(Bg + lo[i], gr[i]);
            cvtStore4(Bu + lo[i], ur[i]);
        }
        __syncthreads();

        // stage tile k+1 while WMMAs run on tile k
        if (k0 + 32 < HID) {
#pragma unroll
            for (int i = 0; i < 4; ++i) {
                ar[i] = *(const uint2*)pa[i];
                gr[i] = *(const float4*)pg[i];
                ur[i] = *(const float4*)pu[i];
                pa[i] += 32; pg[i] += 32; pu[i] += 32;
            }
        }

        v16bf a0 = ldsFrag(As, m0w + lrow,      khalf);
        v16bf a1 = ldsFrag(As, m0w + 16 + lrow, khalf);

        v16bf bg[4];
#pragma unroll
        for (int ni = 0; ni < 4; ++ni) bg[ni] = ldsFrag(Bg, n0w + ni * 16 + lrow, khalf);
#pragma unroll
        for (int ni = 0; ni < 4; ++ni) {
            accg[0][ni] = __builtin_amdgcn_wmma_f32_16x16x32_bf16(false, a0, false, bg[ni], (short)0, accg[0][ni], false, false);
            accg[1][ni] = __builtin_amdgcn_wmma_f32_16x16x32_bf16(false, a1, false, bg[ni], (short)0, accg[1][ni], false, false);
        }
        v16bf bu[4];
#pragma unroll
        for (int ni = 0; ni < 4; ++ni) bu[ni] = ldsFrag(Bu, n0w + ni * 16 + lrow, khalf);
#pragma unroll
        for (int ni = 0; ni < 4; ++ni) {
            accu[0][ni] = __builtin_amdgcn_wmma_f32_16x16x32_bf16(false, a0, false, bu[ni], (short)0, accu[0][ni], false, false);
            accu[1][ni] = __builtin_amdgcn_wmma_f32_16x16x32_bf16(false, a1, false, bu[ni], (short)0, accu[1][ni], false, false);
        }
    }

    const int mlo   = (lane >> 4) << 3;  // C/D: lanes 16-31 hold M = r+8
    const int nlane = lane & 15;
#pragma unroll
    for (int mi = 0; mi < 2; ++mi)
#pragma unroll
        for (int ni = 0; ni < 4; ++ni)
#pragma unroll
            for (int r = 0; r < 8; ++r) {
                int m = m0w + mi * 16 + r + mlo;
                int slot = s0 + m;
                if (slot < c) {
                    float g = accg[mi][ni][r];
                    float u = accu[mi][ni][r];
                    float v = g * (1.0f / (1.0f + __expf(-g))) * u;  // silu(g)*u
                    act[(size_t)(base + slot) * INTER + (n0 + n0w + ni * 16 + nlane)] = (__bf16)v;
                }
            }
}

// ---------------- GEMM2: out += w_tok * (act @ w2^T) ----------------
__global__ __launch_bounds__(256)
void moe_gemm2(const u16* __restrict__ act_u, const float* __restrict__ w2,
               const int* __restrict__ cnt, const int* __restrict__ off,
               const int* __restrict__ tok_ids, const float* __restrict__ tok_w,
               float* __restrict__ out) {
    const int e  = blockIdx.z;
    const int h0 = blockIdx.x * 128;
    const int s0 = blockIdx.y * 128;
    const int c  = cnt[e];
    if (s0 >= c) return;
    const int base = off[e];
    const __bf16* act = (const __bf16*)act_u;

    __shared__ alignas(16) __bf16 As[128 * 40];
    __shared__ alignas(16) __bf16 Bs[128 * 40];
    __shared__ int   ts[128];
    __shared__ float sw[128];
    __shared__ int   ra[128];

    const int tid = threadIdx.x;
    if (tid < 128) {
        int slot = s0 + tid;
        int sc = slot < c ? slot : (c - 1);
        ts[tid] = tok_ids[e * T_TOK + sc];
        sw[tid] = tok_w[e * T_TOK + sc];
        ra[tid] = base + sc;
    }
    __syncthreads();

    const __bf16* pa[4]; const float* pb[4]; int lo[4];
#pragma unroll
    for (int i = 0; i < 4; ++i) {
        int cidx = tid + 256 * i;
        int row  = cidx >> 3;
        int kc   = (cidx & 7) * 4;
        pa[i] = act + (size_t)ra[row] * INTER + kc;
        pb[i] = w2 + ((size_t)e * HID + h0 + row) * INTER + kc;
        lo[i] = row * 40 + kc;
    }

    const int lane  = tid & 31;
    const int wv    = tid >> 5;
    const int m0w   = (wv & 3) * 32;
    const int n0w   = (wv >> 2) * 64;
    const int lrow  = lane & 15;
    const int khalf = (lane >> 4) << 3;

    v8f acc[2][4];
#pragma unroll
    for (int mi = 0; mi < 2; ++mi)
#pragma unroll
        for (int ni = 0; ni < 4; ++ni) acc[mi][ni] = {};

    uint2 ar[4]; float4 br[4];
#pragma unroll
    for (int i = 0; i < 4; ++i) {
        ar[i] = *(const uint2*)pa[i];
        br[i] = *(const float4*)pb[i];
        pa[i] += 32; pb[i] += 32;
    }

    for (int k0 = 0; k0 < INTER; k0 += 32) {
        __syncthreads();
#pragma unroll
        for (int i = 0; i < 4; ++i) {
            *(uint2*)(As + lo[i]) = ar[i];
            cvtStore4(Bs + lo[i], br[i]);
        }
        __syncthreads();

        if (k0 + 32 < INTER) {
#pragma unroll
            for (int i = 0; i < 4; ++i) {
                ar[i] = *(const uint2*)pa[i];
                br[i] = *(const float4*)pb[i];
                pa[i] += 32; pb[i] += 32;
            }
        }

        v16bf a0 = ldsFrag(As, m0w + lrow,      khalf);
        v16bf a1 = ldsFrag(As, m0w + 16 + lrow, khalf);
        v16bf b[4];
#pragma unroll
        for (int ni = 0; ni < 4; ++ni) b[ni] = ldsFrag(Bs, n0w + ni * 16 + lrow, khalf);
#pragma unroll
        for (int ni = 0; ni < 4; ++ni) {
            acc[0][ni] = __builtin_amdgcn_wmma_f32_16x16x32_bf16(false, a0, false, b[ni], (short)0, acc[0][ni], false, false);
            acc[1][ni] = __builtin_amdgcn_wmma_f32_16x16x32_bf16(false, a1, false, b[ni], (short)0, acc[1][ni], false, false);
        }
    }

    const int mlo   = (lane >> 4) << 3;
    const int nlane = lane & 15;
#pragma unroll
    for (int mi = 0; mi < 2; ++mi)
#pragma unroll
        for (int ni = 0; ni < 4; ++ni)
#pragma unroll
            for (int r = 0; r < 8; ++r) {
                int m = m0w + mi * 16 + r + mlo;
                int slot = s0 + m;
                if (slot < c) {
                    float v = acc[mi][ni][r] * sw[m];
                    unsafeAtomicAdd(out + (size_t)ts[m] * HID + (h0 + n0w + ni * 16 + nlane), v);
                }
            }
}

extern "C" void kernel_launch(void* const* d_in, const int* in_sizes, int n_in,
                              void* d_out, int out_size, void* d_ws, size_t ws_size,
                              hipStream_t stream) {
    const float* x   = (const float*)d_in[0];
    const float* rl  = (const float*)d_in[1];
    const float* w13 = (const float*)d_in[2];
    const float* w2  = (const float*)d_in[3];
    float* out = (float*)d_out;

    char* ws = (char*)d_ws;
    size_t o = 0;
    int*   cnt     = (int*)(ws + o);   o += 256;
    int*   off     = (int*)(ws + o);   o += 256;
    int*   tok_ids = (int*)(ws + o);   o += sizeof(int)   * NEXP * T_TOK;
    float* tok_w   = (float*)(ws + o); o += sizeof(float) * NEXP * T_TOK;
    u16*   xb      = (u16*)(ws + o);   o += sizeof(u16) * (size_t)T_TOK * HID;
    u16*   act     = (u16*)(ws + o);   o += sizeof(u16) * (size_t)2 * T_TOK * INTER;

    hipMemsetAsync(d_out, 0, (size_t)T_TOK * HID * sizeof(float), stream);
    hipMemsetAsync(cnt, 0, 256, stream);

    moe_router<<<T_TOK / 256, 256, 0, stream>>>(rl, cnt, tok_ids, tok_w);
    scan_counts<<<1, 32, 0, stream>>>(cnt, off);
    cvt_x<<<(T_TOK * HID / 4) / 256, 256, 0, stream>>>(x, xb, T_TOK * HID / 4);
    moe_gemm1<<<dim3(INTER / 128, T_TOK / 128, NEXP), 256, 0, stream>>>(xb, w13, cnt, off, tok_ids, act);
    moe_gemm2<<<dim3(HID / 128, T_TOK / 128, NEXP), 256, 0, stream>>>(act, w2, cnt, off, tok_ids, tok_w, out);
}
